// GCN_custom_42898133353088
// MI455X (gfx1250) — matrix operand order repaired and checked by reading, not compiled, not run on hardware
//
#include <hip/hip_runtime.h>
#include <hip/hip_bf16.h>

#define B_SZ     2
#define GRID_HW  64800      // 180*360
#define N_NODES  20000
#define HID      512
#define DEPTH    8
#define OUT_DIM  256
#define M_ROWS   (B_SZ * N_NODES)   // 40000, divisible by 32

typedef __attribute__((ext_vector_type(16))) __bf16 v16bf;
typedef __attribute__((ext_vector_type(8)))  float  v8f;

union Frag16 { v16bf v; uint4 q[2]; };

__device__ __forceinline__ unsigned short f2bf(float f) {
    unsigned int u = __float_as_uint(f);
    unsigned int r = u + 0x7FFFu + ((u >> 16) & 1u);   // RNE
    return (unsigned short)(r >> 16);
}

// ---------------------------------------------------------------------------
// Layer-0: T[b,n,h] = sst[b, node_idx[n]] * W1[h]; also stash x0 scalar.
// ---------------------------------------------------------------------------
__global__ __launch_bounds__(256)
void k_init_t1(const float* __restrict__ sst, const int* __restrict__ node_idx,
               const float* __restrict__ W1, float* __restrict__ T,
               float* __restrict__ X0) {
    size_t i = (size_t)blockIdx.x * blockDim.x + threadIdx.x;
    if (i >= (size_t)M_ROWS * HID) return;
    int row = (int)(i >> 9);
    int h   = (int)(i & (HID - 1));
    int b   = row / N_NODES;
    int n   = row - b * N_NODES;
    float x0 = sst[(size_t)b * GRID_HW + node_idx[n]];
    T[i] = x0 * W1[h];
    if (h == 0) X0[row] = x0;
}

// ---------------------------------------------------------------------------
// Zero a float buffer (uint4-wide).
// ---------------------------------------------------------------------------
__global__ __launch_bounds__(256)
void k_zero4(uint4* __restrict__ p, size_t n4) {
    size_t i = (size_t)blockIdx.x * blockDim.x + threadIdx.x;
    if (i < n4) p[i] = make_uint4(0u, 0u, 0u, 0u);
}

// ---------------------------------------------------------------------------
// W [K=512, N=512] f32 -> Wt [N, K] bf16 (K-major so B-fragments are contiguous)
// ---------------------------------------------------------------------------
__global__ __launch_bounds__(256)
void k_transpose_w(const float* __restrict__ W, unsigned short* __restrict__ Wt) {
    int i = blockIdx.x * blockDim.x + threadIdx.x;   // 0 .. 512*512-1
    int n = i >> 9;
    int k = i & (HID - 1);
    Wt[(size_t)n * HID + k] = f2bf(W[(size_t)k * HID + n]);
}

// ---------------------------------------------------------------------------
// GEMM: T[M,512] = Xb[M,512](bf16) @ W (via Wt[N,K] bf16), f32 accumulate.
// 256 threads = 8 waves; block -> 32 rows x 512 cols; wave w -> cols [64w, 64w+64)
// and BOTH 16-row M-subtiles (B fragments reused across them -> half the L2
// B-traffic per FLOP vs a 16-row block). A-tile staged in LDS with rows padded
// to 520 ushort (260 DWORDs -> half-wave ds_load_b128 hits 64 distinct banks).
// B fragments double-buffered in VGPRs: loads for step kk+32 are in flight
// while the 8 WMMAs of step kk execute.
// ---------------------------------------------------------------------------
#define A_PITCH 520   // 512 + 8 ushort pad (16 bytes) per row
#define M_TILE  32

__global__ __launch_bounds__(256)
void k_gemm_bf16(const unsigned short* __restrict__ Xb,
                 const unsigned short* __restrict__ Wt,
                 float* __restrict__ T) {
    __shared__ unsigned short As[M_TILE * A_PITCH];   // 33.3 KB

    const int tid  = threadIdx.x;
    const int lane = tid & 31;
    const int wave = tid >> 5;
    const int m0   = blockIdx.x * M_TILE;
    const int n0w  = wave * 64;
    const bool hi  = (lane >= 16);
    const int  l15 = lane & 15;

    // ---- cooperative A-tile load: 32 rows x 512 bf16 = 2048 uint4 ----
#pragma unroll
    for (int v = 0; v < 8; ++v) {
        int e4  = tid + v * 256;          // uint4 index 0..2047
        int e   = e4 * 8;                 // ushort index
        int row = e >> 9;
        int k   = e & (HID - 1);
        *(uint4*)&As[row * A_PITCH + k] =
            *(const uint4*)(Xb + (size_t)(m0 + row) * HID + k);
    }
    __syncthreads();

    v8f acc[2][4];
#pragma unroll
    for (int m = 0; m < 2; ++m)
#pragma unroll
        for (int t = 0; t < 4; ++t)
            acc[m][t] = (v8f){0.f,0.f,0.f,0.f,0.f,0.f,0.f,0.f};

    // B fragment bases: col n0w + 16t + l15, K = {0..7}/{16..23} (hi half-wave -> +8)
    const unsigned short* bbase[4];
#pragma unroll
    for (int t = 0; t < 4; ++t)
        bbase[t] = Wt + (size_t)(n0w + t * 16 + l15) * HID + (hi ? 8 : 0);

    Frag16 bbuf[2][4];
#pragma unroll
    for (int t = 0; t < 4; ++t) {                 // preload kk = 0
        bbuf[0][t].q[0] = *(const uint4*)(bbase[t]);
        bbuf[0][t].q[1] = *(const uint4*)(bbase[t] + 16);
    }

    // A row pointers for the two 16-row M-subtiles
    const unsigned short* arow0 = &As[l15 * A_PITCH + (hi ? 8 : 0)];
    const unsigned short* arow1 = arow0 + 16 * A_PITCH;

    for (int kk = 0; kk < HID; kk += 32) {
        const int cur = (kk >> 5) & 1;
        const int nxt = cur ^ 1;

        // A fragments from LDS: K = kk+{0..7}/{16..23} (+8 for upper half-wave)
        Frag16 a0, a1;
        a0.q[0] = *(const uint4*)(arow0 + kk);
        a0.q[1] = *(const uint4*)(arow0 + kk + 16);
        a1.q[0] = *(const uint4*)(arow1 + kk);
        a1.q[1] = *(const uint4*)(arow1 + kk + 16);

        if (kk + 32 < HID) {
#pragma unroll
            for (int t = 0; t < 4; ++t) {         // issue next-step B loads first
                bbuf[nxt][t].q[0] = *(const uint4*)(bbase[t] + kk + 32);
                bbuf[nxt][t].q[1] = *(const uint4*)(bbase[t] + kk + 48);
            }
            if (kk + 64 < HID)
                __builtin_prefetch((const void*)(bbase[0] + kk + 64), 0, 1);
        }

#pragma unroll
        for (int t = 0; t < 4; ++t) {
            acc[0][t] = __builtin_amdgcn_wmma_f32_16x16x32_bf16(
                false, a0.v, false, bbuf[cur][t].v, (short)0, acc[0][t], false, false);
            acc[1][t] = __builtin_amdgcn_wmma_f32_16x16x32_bf16(
                false, a1.v, false, bbuf[cur][t].v, (short)0, acc[1][t], false, false);
        }
    }

    // C/D layout: element i -> row m0 + 16m + i + (hi?8:0), col n0w + 16t + l15
#pragma unroll
    for (int m = 0; m < 2; ++m) {
#pragma unroll
        for (int t = 0; t < 4; ++t) {
            float* out = T + (size_t)(m0 + m * 16 + (hi ? 8 : 0)) * HID
                           + n0w + t * 16 + l15;
#pragma unroll
            for (int i = 0; i < 8; ++i) out[(size_t)i * HID] = acc[m][t][i];
        }
    }
}

// ---------------------------------------------------------------------------
// SpMM scatter: Y[b, erow[e], :] += evals[e] * S[b, ecol[e], :]
// One block per edge, 128 threads x float4 over HID, loop over batch.
// ---------------------------------------------------------------------------
__global__ __launch_bounds__(128)
void k_spmm(const int* __restrict__ erow, const int* __restrict__ ecol,
            const float* __restrict__ ev, const float* __restrict__ S,
            float* __restrict__ Y, int nE) {
    int e = blockIdx.x;
    if (e >= nE) return;
    int r = erow[e], c = ecol[e];
    float v = ev[e];
    int h = threadIdx.x * 4;
#pragma unroll
    for (int b = 0; b < B_SZ; ++b) {
        const float4 s = *(const float4*)(S + ((size_t)b * N_NODES + c) * HID + h);
        float* y = Y + ((size_t)b * N_NODES + r) * HID + h;
        __hip_atomic_fetch_add(y + 0, s.x * v, __ATOMIC_RELAXED, __HIP_MEMORY_SCOPE_AGENT);
        __hip_atomic_fetch_add(y + 1, s.y * v, __ATOMIC_RELAXED, __HIP_MEMORY_SCOPE_AGENT);
        __hip_atomic_fetch_add(y + 2, s.z * v, __ATOMIC_RELAXED, __HIP_MEMORY_SCOPE_AGENT);
        __hip_atomic_fetch_add(y + 3, s.w * v, __ATOMIC_RELAXED, __HIP_MEMORY_SCOPE_AGENT);
    }
}

// ---------------------------------------------------------------------------
// Residual + LeakyReLU; also refresh the bf16 shadow copy for the next GEMM.
// first=1: base = X0[row] (broadcast scalar); else base = X[i].
// ---------------------------------------------------------------------------
__global__ __launch_bounds__(256)
void k_resid(const float* __restrict__ Y, const float* __restrict__ bias,
             float* __restrict__ X, unsigned short* __restrict__ Xb,
             const float* __restrict__ X0, int first) {
    size_t i = (size_t)blockIdx.x * blockDim.x + threadIdx.x;
    if (i >= (size_t)M_ROWS * HID) return;
    int h = (int)(i & (HID - 1));
    float base = first ? X0[i >> 9] : X[i];
    float v = Y[i] + bias[h];
    v = (v >= 0.f) ? v : 0.01f * v;
    float o = base + v;
    X[i]  = o;
    Xb[i] = f2bf(o);
}

// ---------------------------------------------------------------------------
// Mean over nodes: one block per (b,h), LDS tree reduce.
// ---------------------------------------------------------------------------
__global__ __launch_bounds__(256)
void k_mean(const float* __restrict__ X, float* __restrict__ M) {
    int b = blockIdx.x >> 9;
    int h = blockIdx.x & (HID - 1);
    __shared__ float red[256];
    float s = 0.f;
    for (int n = threadIdx.x; n < N_NODES; n += 256)
        s += X[((size_t)b * N_NODES + n) * HID + h];
    red[threadIdx.x] = s;
    __syncthreads();
    for (int off = 128; off > 0; off >>= 1) {
        if ((int)threadIdx.x < off) red[threadIdx.x] += red[threadIdx.x + off];
        __syncthreads();
    }
    if (threadIdx.x == 0) M[b * HID + h] = red[0] * (1.f / N_NODES);
}

// ---------------------------------------------------------------------------
// Head: out[b,o] = dot(M[b,:], W_head[o,:]) + b_head[o]
// ---------------------------------------------------------------------------
__global__ __launch_bounds__(256)
void k_head(const float* __restrict__ M, const float* __restrict__ Wh,
            const float* __restrict__ bh, float* __restrict__ out) {
    int idx = blockIdx.x * blockDim.x + threadIdx.x;
    if (idx >= B_SZ * OUT_DIM) return;
    int b = idx >> 8, o = idx & (OUT_DIM - 1);
    float s = bh[o];
    for (int h = 0; h < HID; ++h) s += M[b * HID + h] * Wh[(size_t)o * HID + h];
    out[idx] = s;
}

// ---------------------------------------------------------------------------
extern "C" void kernel_launch(void* const* d_in, const int* in_sizes, int n_in,
                              void* d_out, int out_size, void* d_ws, size_t ws_size,
                              hipStream_t stream) {
    const float* sst      = (const float*)d_in[0];
    const int*   node_idx = (const int*)  d_in[1];
    const int*   erow     = (const int*)  d_in[2];
    const int*   ecol     = (const int*)  d_in[3];
    const float* evals    = (const float*)d_in[4];
    const float* W1       = (const float*)d_in[5];
    const float* b1       = (const float*)d_in[6];
    const float* Ws       = (const float*)d_in[7];   // [DEPTH,512,512]
    const float* bs       = (const float*)d_in[8];   // [DEPTH,512]
    const float* W_head   = (const float*)d_in[9];
    const float* b_head   = (const float*)d_in[10];
    const int nE = in_sizes[2];

    // Workspace carve-up (all 256B aligned).
    const size_t elems = (size_t)M_ROWS * HID;           // 20,480,000
    char* p = (char*)d_ws;
    auto take = [&](size_t bytes) { char* q = p; p += (bytes + 255) & ~(size_t)255; return q; };
    float*          X    = (float*)         take(elems * sizeof(float));
    float*          T    = (float*)         take(elems * sizeof(float));
    float*          Y    = (float*)         take(elems * sizeof(float));
    unsigned short* Xb   = (unsigned short*)take(elems * sizeof(unsigned short));
    unsigned short* Wt   = (unsigned short*)take((size_t)HID * HID * sizeof(unsigned short));
    float*          X0   = (float*)         take((size_t)M_ROWS * sizeof(float));
    float*          Mean = (float*)         take((size_t)B_SZ * HID * sizeof(float));
    (void)ws_size; (void)n_in;

    const int eltGrid  = (int)((elems + 255) / 256);
    const size_t n4    = elems / 4;
    const int zeroGrid = (int)((n4 + 255) / 256);

    // ---- layer 0 (conv1: 1 -> HID, outer product) ----
    k_init_t1<<<eltGrid, 256, 0, stream>>>(sst, node_idx, W1, T, X0);
    k_zero4 <<<zeroGrid, 256, 0, stream>>>((uint4*)Y, n4);
    k_spmm  <<<nE, 128, 0, stream>>>(erow, ecol, evals, T, Y, nE);
    k_resid <<<eltGrid, 256, 0, stream>>>(Y, b1, X, Xb, X0, 1);

    // ---- layers 1..DEPTH (HID -> HID, bf16 WMMA GEMM) ----
    for (int d = 0; d < DEPTH; ++d) {
        k_transpose_w<<<(HID * HID) / 256, 256, 0, stream>>>(Ws + (size_t)d * HID * HID, Wt);
        k_gemm_bf16  <<<M_ROWS / M_TILE, 256, 0, stream>>>(Xb, Wt, T);
        k_zero4      <<<zeroGrid, 256, 0, stream>>>((uint4*)Y, n4);
        k_spmm       <<<nE, 128, 0, stream>>>(erow, ecol, evals, T, Y, nE);
        k_resid      <<<eltGrid, 256, 0, stream>>>(Y, bs + (size_t)d * HID, X, Xb, X0, 0);
    }

    // ---- mean over nodes + head ----
    k_mean<<<B_SZ * HID, 256, 0, stream>>>(X, Mean);
    k_head<<<(B_SZ * OUT_DIM + 255) / 256, 256, 0, stream>>>(Mean, W_head, b_head, (float*)d_out);
    (void)out_size;
}
